// DGCNcell_27951647162914
// MI455X (gfx1250) — matrix operand order repaired and checked by reading, not compiled, not run on hardware
//
#include <hip/hip_runtime.h>
#include <hip/hip_bf16.h>

// MI455X / gfx1250, wave32. All GEMMs via v_wmma_f32_16x16x32_f16 (fp32 accum).
// All WMMA operands are PRE-PACKED in fragment order so each lane loads its 16
// f16 elements with contiguous b128 loads. Two 16-row m-tiles per wave to
// double the WMMA:load ratio and halve L2 operand re-traffic.
typedef __attribute__((ext_vector_type(16))) _Float16 v16h;
typedef __attribute__((ext_vector_type(8)))  float    v8f;

#define B_  8
#define N_  2048
#define NT16 (N_ / 16)   // 128 tiles of 16
#define NT32 (N_ / 32)   // 64 tiles of 32
#define NEGC (-1e16f)

// ---- fragment-order offsets within one 32(K)x16(M) tile of 512 elems -------
// A-frag (CDNA5 ISA 7.12.2, 16-bit A 16x32): lane = M + 16*((K>>3)&1),
//   elem e = (K&7) + 8*(K>>4)
__host__ __device__ __forceinline__ int a_off(int K, int M) {
  return (M + (((K >> 3) & 1) << 4)) * 16 + (K & 7) + ((K >> 4) << 3);
}
// B-frag (32x16): lane = M + 16*((K>>3)&1), elem e = (K&7) + 8*(K>>4)
__host__ __device__ __forceinline__ int b_off(int K, int M) {
  return (M + (((K >> 3) & 1) << 4)) * 16 + (K & 7) + ((K >> 4) << 3);
}
// D/C-frag (f32 16x16): lane = col + 16*(row>>3), reg r = row&7
__host__ __device__ __forceinline__ int d_off(int row, int col) {
  return (col + ((row >> 3) << 4)) * 8 + (row & 7);
}

__device__ __forceinline__ v8f wmma_f16(v16h a, v16h b, v8f c) {
  return __builtin_amdgcn_wmma_f32_16x16x32_f16(false, a, false, b, (short)0, c,
                                                false, false);
}
__device__ __forceinline__ v16h ldfrag(const _Float16* p) {
  return *(const v16h*)p;
}

// ---- prep: x = input@linin_W+b; gP = [x,state] packed A-frags; demand; -----
__global__ void prep_kernel(const float* __restrict__ input,
                            const float* __restrict__ state,
                            const float* __restrict__ convW,
                            const float* __restrict__ convb,
                            const float* __restrict__ lininW,
                            const float* __restrict__ lininb,
                            _Float16* __restrict__ gP,
                            float* __restrict__ xbuf,
                            float* __restrict__ demand,
                            float* __restrict__ inpP) {
  int idx = blockIdx.x * blockDim.x + threadIdx.x;
  if (idx >= B_ * N_) return;
  const int b = idx / N_, n = idx % N_;
  float i0 = input[idx * 2], i1 = input[idx * 2 + 1];
  inpP[(0 * B_ + b) * N_ + n] = i0;
  inpP[(1 * B_ + b) * N_ + n] = i1;
  float g[64];
#pragma unroll
  for (int h = 0; h < 32; ++h) {
    float x = lininb[h] + i0 * lininW[h] + i1 * lininW[32 + h];
    g[h] = x;
    xbuf[idx * 32 + h] = x;
  }
#pragma unroll
  for (int h = 0; h < 32; ++h) g[32 + h] = state[idx * 32 + h];
  const int tn = n >> 5, K = n & 31;
  _Float16* base = gP + (((long)b * NT32 + tn) * 4) * 512;
#pragma unroll
  for (int f = 0; f < 64; ++f)
    base[(f >> 4) * 512 + a_off(K, f & 15)] = (_Float16)g[f];
#pragma unroll
  for (int c = 0; c < 2; ++c) {
    float d = convb[c];
#pragma unroll
    for (int f = 0; f < 64; ++f) d += g[f] * convW[f * 2 + c];
    demand[idx * 2 + c] = tanhf(d) * 0.5f;
  }
}

// ---- one-time conversion of N x N operands into packed f16 B-frag tiles ----
__global__ void cvt_weights_kernel(const float* __restrict__ A,
                                   const float* __restrict__ Badj,
                                   const float* __restrict__ w2c,
                                   const float* __restrict__ wr,
                                   const float* __restrict__ wu,
                                   const float* __restrict__ wc,
                                   _Float16* WA, _Float16* WB, _Float16* WR,
                                   _Float16* WU, _Float16* WC, float* MSKA,
                                   float* MSKB) {
  long i = (long)blockIdx.x * blockDim.x + threadIdx.x;
  if (i >= (long)N_ * N_) return;
  const int n = (int)(i / N_), m = (int)(i % N_);
  float a = A[i], bd = Badj[i];
  long wi = (((long)(n >> 5) * NT16 + (m >> 4)) << 9) + b_off(n & 31, m & 15);
  WA[wi] = (_Float16)(a * w2c[i]);
  WB[wi] = (_Float16)(bd * w2c[(long)N_ * N_ + i]);
  WR[wi] = (_Float16)(a * wr[i]);
  WU[wi] = (_Float16)(a * wu[i]);
  WC[wi] = (_Float16)(a * wc[i]);
  long di = (((long)(n >> 4) * NT16 + (m >> 4)) << 8) + d_off(n & 15, m & 15);
  MSKA[di] = NEGC * (1.0f - a);
  MSKB[di] = NEGC * (1.0f - bd);
}

// ---- small conversions: w1c -> packed B-frags; Lin^T -> packed A-frags -----
__global__ void cvt_small_kernel(const float* __restrict__ w1c,
                                 const float* __restrict__ linrW,
                                 const float* __restrict__ linuW,
                                 const float* __restrict__ lincW,
                                 const float* __restrict__ br,
                                 const float* __restrict__ bu,
                                 const float* __restrict__ bc, _Float16* W1P,
                                 _Float16* linrPk, _Float16* linuPk,
                                 _Float16* lincPk, float* brp, float* bup,
                                 float* bcp) {
  int i = blockIdx.x * blockDim.x + threadIdx.x;
  if (i < 2 * 64 * N_) {
    int at = i / (64 * N_), f = (i / N_) % 64, k = i % N_;
    long wi = ((((long)at * 2 + (f >> 5)) * NT16 + (k >> 4)) << 9) +
              b_off(f & 31, k & 15);
    W1P[wi] = (_Float16)w1c[i];
  }
  if (i < 64 * 32) {
    int d = i / 32, h = i % 32;
    int idx = (((h >> 4) * 2 + (d >> 5)) << 9) + a_off(d & 31, h & 15);
    linrPk[idx] = (_Float16)((d < 34) ? linrW[d * 32 + h] : 0.f);
    linuPk[idx] = (_Float16)((d < 34) ? linuW[d * 32 + h] : 0.f);
    lincPk[idx] = (_Float16)lincW[i];
  }
  if (i < 64) {
    brp[i] = (i < 34) ? br[i] : 0.f;
    bup[i] = (i < 34) ? bu[i] : 0.f;
    bcp[i] = bc[i];
  }
}

// ---- fused attention branch: 2 m-tiles (32 rows) per wave ------------------
__global__ void __launch_bounds__(32) attn_fused_kernel(
    const _Float16* __restrict__ gP, const _Float16* __restrict__ WAp,
    const _Float16* __restrict__ WBp, const float* __restrict__ MSKA,
    const float* __restrict__ MSKB, const _Float16* __restrict__ W1P,
    const float* __restrict__ inpP, const float* __restrict__ demand,
    float* __restrict__ p_out, _Float16* __restrict__ fruP) {
  __shared__ __align__(32) _Float16 hA[2 * 1024];  // h A-frags, 2 m-tiles
  const unsigned lane = threadIdx.x;
  const int mtA = blockIdx.x * 2, mtB = mtA + 1;
  const int b  = blockIdx.y;
  const int at = blockIdx.z;
  const _Float16* Wp  = at ? WBp : WAp;
  const float*    MSK = at ? MSKB : MSKA;
  const int le = lane << 4;

  // Stage 1: h^T = G^T * W for both m-tiles (8 accumulators, 8 WMMA/iter).
  v8f acc[2][4];
#pragma unroll
  for (int t = 0; t < 2; ++t)
#pragma unroll
    for (int j = 0; j < 4; ++j) acc[t][j] = (v8f){};
  for (int tn = 0; tn < NT32; ++tn) {
    const _Float16* gt = gP + ((((long)b * NT32 + tn) * 4) << 9) + le;
    v16h a0 = ldfrag(gt + 0 * 512);
    v16h a1 = ldfrag(gt + 1 * 512);
    v16h a2 = ldfrag(gt + 2 * 512);
    v16h a3 = ldfrag(gt + 3 * 512);
    v16h bf0 = ldfrag(Wp + (((long)tn * NT16 + mtA) << 9) + le);
    v16h bf1 = ldfrag(Wp + (((long)tn * NT16 + mtB) << 9) + le);
    acc[0][0] = wmma_f16(a0, bf0, acc[0][0]);
    acc[0][1] = wmma_f16(a1, bf0, acc[0][1]);
    acc[0][2] = wmma_f16(a2, bf0, acc[0][2]);
    acc[0][3] = wmma_f16(a3, bf0, acc[0][3]);
    acc[1][0] = wmma_f16(a0, bf1, acc[1][0]);
    acc[1][1] = wmma_f16(a1, bf1, acc[1][1]);
    acc[1][2] = wmma_f16(a2, bf1, acc[1][2]);
    acc[1][3] = wmma_f16(a3, bf1, acc[1][3]);
  }
  const int col    = lane & 15u;
  const int rowOff = (lane >> 4) << 3;
#pragma unroll
  for (int t = 0; t < 2; ++t)
#pragma unroll
    for (int j = 0; j < 4; ++j)
#pragma unroll
      for (int r = 0; r < 8; ++r) {
        int f = j * 16 + r + rowOff;
        hA[t * 1024 + (f >> 5) * 512 + a_off(f & 31, col)] =
            (_Float16)acc[t][j][r];
      }
  __syncthreads();
  v16h ha[2][2];
#pragma unroll
  for (int t = 0; t < 2; ++t) {
    ha[t][0] = ldfrag(hA + t * 1024 + 0 * 512 + le);
    ha[t][1] = ldfrag(hA + t * 1024 + 1 * 512 + le);
  }

  // Stage 2: flash masked softmax over 128 k-tiles, both m-tiles share W1/inp.
  float mrun[2][8], lrun[2][8], vrun[2][8];
#pragma unroll
  for (int t = 0; t < 2; ++t)
#pragma unroll
    for (int r = 0; r < 8; ++r) {
      mrun[t][r] = -INFINITY;
      lrun[t][r] = 0.f;
      vrun[t][r] = 0.f;
    }
  const float* inpC = inpP + ((long)at * B_ + b) * N_;
  for (int kt = 0; kt < NT16; ++kt) {
    v16h wb0 = ldfrag(W1P + ((((long)at * 2 + 0) * NT16 + kt) << 9) + le);
    v16h wb1 = ldfrag(W1P + ((((long)at * 2 + 1) * NT16 + kt) << 9) + le);
    v8f s0 = {}, s1 = {};
    s0 = wmma_f16(ha[0][0], wb0, s0);
    s0 = wmma_f16(ha[0][1], wb1, s0);
    s1 = wmma_f16(ha[1][0], wb0, s1);
    s1 = wmma_f16(ha[1][1], wb1, s1);
    v8f msk0 = *(const v8f*)(MSK + ((((long)mtA * NT16 + kt) << 5) + lane) * 8);
    v8f msk1 = *(const v8f*)(MSK + ((((long)mtB * NT16 + kt) << 5) + lane) * 8);
    float cinp = inpC[kt * 16 + col];
#pragma unroll
    for (int r = 0; r < 8; ++r) {
      float sv0 = s0[r] + msk0[r];
      float nm0 = fmaxf(mrun[0][r], sv0);
      float sc0 = __expf(mrun[0][r] - nm0);
      float e00 = __expf(sv0 - nm0);
      lrun[0][r] = lrun[0][r] * sc0 + e00;
      vrun[0][r] = vrun[0][r] * sc0 + e00 * cinp;
      mrun[0][r] = nm0;
      float sv1 = s1[r] + msk1[r];
      float nm1 = fmaxf(mrun[1][r], sv1);
      float sc1 = __expf(mrun[1][r] - nm1);
      float e01 = __expf(sv1 - nm1);
      lrun[1][r] = lrun[1][r] * sc1 + e01;
      vrun[1][r] = vrun[1][r] * sc1 + e01 * cinp;
      mrun[1][r] = nm1;
    }
  }
  // Merge 16 lanes per half (xor<16 keeps halves independent under wave32).
#pragma unroll
  for (int t = 0; t < 2; ++t)
#pragma unroll
    for (int r = 0; r < 8; ++r) {
      float M = mrun[t][r];
      for (int d = 1; d < 16; d <<= 1) M = fmaxf(M, __shfl_xor(M, d, 32));
      float l = lrun[t][r] * __expf(mrun[t][r] - M);
      float v = vrun[t][r] * __expf(mrun[t][r] - M);
      for (int d = 1; d < 16; d <<= 1) {
        l += __shfl_xor(l, d, 32);
        v += __shfl_xor(v, d, 32);
      }
      if (col == 0) {
        int row = (mtA + t) * 16 + r + rowOff;
        long o = ((long)b * N_ + row) * 2 + at;
        float pv = v / l + demand[o];
        p_out[o] = pv;
        fruP[((((long)b * NT32 + (row >> 5)) * 4) << 9) + a_off(row & 31, at)] =
            (_Float16)pv;
      }
    }
}

// ---- LocalGC (W^T G) + small linear (WMMA) + activation, 2 m-tiles ---------
__global__ void __launch_bounds__(32) gc_kernel(
    const _Float16* __restrict__ Wp, const _Float16* __restrict__ Gp,
    const float* __restrict__ biasD, const _Float16* __restrict__ LinPk,
    const float* __restrict__ linb, float* __restrict__ out, int act) {
  __shared__ __align__(32) _Float16 tB[2 * 1024];  // tmp B-frags, 2 m-tiles
  const unsigned lane = threadIdx.x;
  const int mtA = blockIdx.x * 2, mtB = mtA + 1;
  const int b  = blockIdx.y;
  const int le = lane << 4;

  v8f acc[2][4];
#pragma unroll
  for (int t = 0; t < 2; ++t)
#pragma unroll
    for (int j = 0; j < 4; ++j) acc[t][j] = (v8f){};
  for (int tn = 0; tn < NT32; ++tn) {
    const _Float16* gt = Gp + ((((long)b * NT32 + tn) * 4) << 9) + le;
    v16h a0 = ldfrag(gt + 0 * 512);
    v16h a1 = ldfrag(gt + 1 * 512);
    v16h a2 = ldfrag(gt + 2 * 512);
    v16h a3 = ldfrag(gt + 3 * 512);
    v16h bf0 = ldfrag(Wp + (((long)tn * NT16 + mtA) << 9) + le);
    v16h bf1 = ldfrag(Wp + (((long)tn * NT16 + mtB) << 9) + le);
    acc[0][0] = wmma_f16(a0, bf0, acc[0][0]);
    acc[0][1] = wmma_f16(a1, bf0, acc[0][1]);
    acc[0][2] = wmma_f16(a2, bf0, acc[0][2]);
    acc[0][3] = wmma_f16(a3, bf0, acc[0][3]);
    acc[1][0] = wmma_f16(a0, bf1, acc[1][0]);
    acc[1][1] = wmma_f16(a1, bf1, acc[1][1]);
    acc[1][2] = wmma_f16(a2, bf1, acc[1][2]);
    acc[1][3] = wmma_f16(a3, bf1, acc[1][3]);
  }
  const int col    = lane & 15u;
  const int rowOff = (lane >> 4) << 3;
#pragma unroll
  for (int t = 0; t < 2; ++t)
#pragma unroll
    for (int j = 0; j < 4; ++j)
#pragma unroll
      for (int r = 0; r < 8; ++r) {
        int d = j * 16 + r + rowOff;
        tB[t * 1024 + (d >> 5) * 512 + b_off(d & 31, col)] =
            (_Float16)(acc[t][j][r] + biasD[d]);
      }
  __syncthreads();
  // out^T = Lin^T (32x64) x tmp (64x16) : Lin frags shared by both m-tiles.
  v16h l00 = ldfrag(LinPk + 0 * 512 + le);
  v16h l01 = ldfrag(LinPk + 1 * 512 + le);
  v16h l10 = ldfrag(LinPk + 2 * 512 + le);
  v16h l11 = ldfrag(LinPk + 3 * 512 + le);
#pragma unroll
  for (int t = 0; t < 2; ++t) {
    v16h b0 = ldfrag(tB + t * 1024 + 0 * 512 + le);
    v16h b1 = ldfrag(tB + t * 1024 + 1 * 512 + le);
    v8f o0 = {}, o1 = {};
    o0 = wmma_f16(l00, b0, o0);
    o0 = wmma_f16(l01, b1, o0);
    o1 = wmma_f16(l10, b0, o1);
    o1 = wmma_f16(l11, b1, o1);
    float* ob = out + ((long)b * N_ + (mtA + t) * 16 + col) * 32;
#pragma unroll
    for (int r = 0; r < 8; ++r) {
      int h = r + rowOff;
      float s0 = o0[r] + linb[h];
      float s1 = o1[r] + linb[16 + h];
      s0 = act ? tanhf(s0) : 1.f / (1.f + __expf(-s0));
      s1 = act ? tanhf(s1) : 1.f / (1.f + __expf(-s1));
      ob[h] = s0;
      ob[16 + h] = s1;
    }
  }
}

// ---- glue: build packed activations ---------------------------------------
__global__ void build_fru_kernel(const float* __restrict__ state,
                                 _Float16* __restrict__ fruP) {
  int idx = blockIdx.x * blockDim.x + threadIdx.x;
  if (idx >= B_ * N_) return;
  const int b = idx / N_, n = idx % N_;
  const int tn = n >> 5, K = n & 31;
  _Float16* base = fruP + ((((long)b * NT32 + tn) * 4) << 9);
  // dims 0/1 (p) are written by attn_fused_kernel; fill 2..33 = state, rest 0
#pragma unroll
  for (int d = 2; d < 64; ++d)
    base[(d >> 4) * 512 + a_off(K, d & 15)] =
        (_Float16)((d < 34) ? state[idx * 32 + (d - 2)] : 0.f);
}

__global__ void build_fc_kernel(const float* __restrict__ xbuf,
                                const float* __restrict__ r,
                                const float* __restrict__ state,
                                _Float16* __restrict__ fcP) {
  int idx = blockIdx.x * blockDim.x + threadIdx.x;
  if (idx >= B_ * N_) return;
  const int b = idx / N_, n = idx % N_;
  const int tn = n >> 5, K = n & 31;
  _Float16* base = fcP + ((((long)b * NT32 + tn) * 4) << 9);
#pragma unroll
  for (int d = 0; d < 64; ++d) {
    float v = (d < 32) ? xbuf[idx * 32 + d]
                       : r[idx * 32 + d - 32] * state[idx * 32 + d - 32];
    base[(d >> 4) * 512 + a_off(K, d & 15)] = (_Float16)v;
  }
}

__global__ void final_h_kernel(const float* __restrict__ u,
                               const float* __restrict__ state,
                               const float* __restrict__ c,
                               float* __restrict__ H) {
  int i = blockIdx.x * blockDim.x + threadIdx.x;
  if (i >= B_ * N_ * 32) return;
  float uv = u[i];
  H[i] = uv * state[i] + (1.f - uv) * c[i];
}

extern "C" void kernel_launch(void* const* d_in, const int* in_sizes, int n_in,
                              void* d_out, int out_size, void* d_ws,
                              size_t ws_size, hipStream_t stream) {
  const float* input  = (const float*)d_in[0];
  const float* state  = (const float*)d_in[1];
  const float* A      = (const float*)d_in[2];
  const float* Badj   = (const float*)d_in[3];
  const float* w1c    = (const float*)d_in[4];
  const float* w2c    = (const float*)d_in[5];
  // d_in[6] = biasc: per-row bias inside softmax -> softmax-invariant, unused.
  const float* convW  = (const float*)d_in[7];
  const float* convb  = (const float*)d_in[8];
  const float* lininW = (const float*)d_in[9];
  const float* lininb = (const float*)d_in[10];
  const float* wr     = (const float*)d_in[11];
  const float* br     = (const float*)d_in[12];
  const float* linrW  = (const float*)d_in[13];
  const float* linrb  = (const float*)d_in[14];
  const float* wu     = (const float*)d_in[15];
  const float* bu     = (const float*)d_in[16];
  const float* linuW  = (const float*)d_in[17];
  const float* linub  = (const float*)d_in[18];
  const float* wc     = (const float*)d_in[19];
  const float* bc     = (const float*)d_in[20];
  const float* lincW  = (const float*)d_in[21];
  const float* lincb  = (const float*)d_in[22];

  char* ws = (char*)d_ws;
  size_t off = 0;
  auto alloc = [&](size_t bytes) {
    void* p = ws + off;
    off = (off + bytes + 255) & ~(size_t)255;
    return p;
  };
  const long NN = (long)N_ * N_;
  _Float16* gP    = (_Float16*)alloc((size_t)B_ * N_ * 64 * 2);
  float*    xbuf  = (float*)alloc((size_t)B_ * N_ * 32 * 4);
  float*    dem   = (float*)alloc((size_t)B_ * N_ * 2 * 4);
  float*    inpP  = (float*)alloc((size_t)2 * B_ * N_ * 4);
  _Float16* WAp   = (_Float16*)alloc(NN * 2);
  _Float16* WBp   = (_Float16*)alloc(NN * 2);
  _Float16* WRp   = (_Float16*)alloc(NN * 2);
  _Float16* WUp   = (_Float16*)alloc(NN * 2);
  _Float16* WCp   = (_Float16*)alloc(NN * 2);
  float*    MSKA  = (float*)alloc(NN * 4);
  float*    MSKB  = (float*)alloc(NN * 4);
  _Float16* W1P   = (_Float16*)alloc((size_t)2 * 64 * N_ * 2);
  _Float16* linrPk = (_Float16*)alloc(4 * 512 * 2);
  _Float16* linuPk = (_Float16*)alloc(4 * 512 * 2);
  _Float16* lincPk = (_Float16*)alloc(4 * 512 * 2);
  float*    brp   = (float*)alloc(256);
  float*    bup   = (float*)alloc(256);
  float*    bcp   = (float*)alloc(256);
  float*    rbuf  = (float*)alloc((size_t)B_ * N_ * 32 * 4);
  float*    ubuf  = (float*)alloc((size_t)B_ * N_ * 32 * 4);
  float*    cbuf  = (float*)alloc((size_t)B_ * N_ * 32 * 4);
  _Float16* fruP  = (_Float16*)alloc((size_t)B_ * N_ * 64 * 2);
  _Float16* fcP   = (_Float16*)alloc((size_t)B_ * N_ * 64 * 2);
  (void)ws_size;  // total ~92 MB

  float* p_out = (float*)d_out;                        // [B,N,2]
  float* H_out = (float*)d_out + (size_t)B_ * N_ * 2;  // [B,N,32]

  const int bn = B_ * N_;
  prep_kernel<<<(bn + 255) / 256, 256, 0, stream>>>(
      input, state, convW, convb, lininW, lininb, gP, xbuf, dem, inpP);
  cvt_weights_kernel<<<(int)((NN + 255) / 256), 256, 0, stream>>>(
      A, Badj, w2c, wr, wu, wc, WAp, WBp, WRp, WUp, WCp, MSKA, MSKB);
  cvt_small_kernel<<<(2 * 64 * N_ + 255) / 256, 256, 0, stream>>>(
      w1c, linrW, linuW, lincW, br, bu, bc, W1P, linrPk, linuPk, lincPk, brp,
      bup, bcp);
  build_fru_kernel<<<(bn + 255) / 256, 256, 0, stream>>>(state, fruP);
  attn_fused_kernel<<<dim3(NT16 / 2, B_, 2), 32, 0, stream>>>(
      gP, WAp, WBp, MSKA, MSKB, W1P, inpP, dem, p_out, fruP);
  gc_kernel<<<dim3(NT16 / 2, B_), 32, 0, stream>>>(WRp, fruP, brp, linrPk,
                                                   linrb, rbuf, 0);
  gc_kernel<<<dim3(NT16 / 2, B_), 32, 0, stream>>>(WUp, fruP, bup, linuPk,
                                                   linub, ubuf, 0);
  build_fc_kernel<<<(bn + 255) / 256, 256, 0, stream>>>(xbuf, rbuf, state, fcP);
  gc_kernel<<<dim3(NT16 / 2, B_), 32, 0, stream>>>(WCp, fcP, bcp, lincPk,
                                                   lincb, cbuf, 1);
  final_h_kernel<<<(bn * 32 + 255) / 256, 256, 0, stream>>>(ubuf, state, cbuf,
                                                            H_out);
}
